// InvariantFFHist_23974507446625
// MI455X (gfx1250) — matrix operand order repaired and checked by reading, not compiled, not run on hardware
//
#include <hip/hip_runtime.h>
#include <math.h>

#define U1C 101
#define TC  200
#define BC  64
#define BTC (BC * TC)          // 12800
#define NROWS (BTC * U1C)      // 1292800
#define WAVES 8
#define NT 7                   // 7 * 16 = 112 >= 100 output cols
#define KS 4                   // 4 * 32 = 128 >= 100 K
#define HSTRIDE 130            // odd-dword stride: avoids LDS bank conflicts

// fragment-ordered weight blob (halfs): W1f | W2f | W3f | W4f
#define W1F_OFF 0              // NT*32*16          = 3584
#define W2F_OFF 3584           // KS*NT*32*16       = 14336
#define W3F_OFF 17920
#define W4F_OFF 32256          // KS*32*16          = 2048
#define WBLOB_HALFS 34304      // total halfs (68,608 bytes)
#define WBLOB_B128 4288        // 16-byte chunks
#define BIAS_FLOATS 336        // 3 x 112

typedef __attribute__((ext_vector_type(16))) _Float16 v16h;
typedef __attribute__((ext_vector_type(8)))  float    v8f;
typedef __attribute__((__vector_size__(16))) int      v4i;

__device__ __forceinline__ v8f wmma16(v16h a, v16h b, v8f c) {
  return __builtin_amdgcn_wmma_f32_16x16x32_f16(false, a, false, b, (short)0, c,
                                                false, false);
}

// 16-bit B-matrix 32x16 layout: lanes 0-15 hold K=0..7,16..23; lanes 16-31 K=8..15,24..31
__device__ __forceinline__ int kmapB(int lane, int j) {
  return (lane < 16 ? 0 : 8) + (j < 8 ? j : 8 + j);
}

// ---------------------------------------------------------------- mean_w ----
__global__ __launch_bounds__(256)
void meanw_kernel(const float* __restrict__ w, float* __restrict__ mw) {
  int row  = blockIdx.x * 8 + (threadIdx.x >> 5);   // BTC rows
  int lane = threadIdx.x & 31;
  const float* p = w + (size_t)row * U1C;
  float s = 0.f;
  for (int j = lane; j < U1C; j += 32) s += p[j];
  for (int off = 16; off > 0; off >>= 1) s += __shfl_xor(s, off, 32);
  if (lane == 0) mw[row] = s * (1.f / (float)U1C);
}

// ---------------------- one-shot f32 -> f16 fragment-order weight prep ------
__global__ __launch_bounds__(256)
void prep_weights_kernel(const float* __restrict__ W1, const float* __restrict__ W2,
                         const float* __restrict__ W3, const float* __restrict__ W4,
                         const float* __restrict__ b1, const float* __restrict__ b2,
                         const float* __restrict__ b3,
                         _Float16* __restrict__ wblob, float* __restrict__ biasbuf) {
  const int e = blockIdx.x * 256 + threadIdx.x;     // 0..WBLOB_HALFS-1
  float v;
  if (e < W2F_OFF) {                                // W1f [nt][lane][16]
    int j = e & 15, ln = (e >> 4) & 31, nt = e >> 9;
    int k = kmapB(ln, j), n = nt * 16 + (ln & 15);
    v = (k < 13 && n < 100) ? W1[k * 100 + n] : 0.f;
  } else if (e < W3F_OFF) {                         // W2f [s][nt][lane][16]
    int e2 = e - W2F_OFF;
    int j = e2 & 15, ln = (e2 >> 4) & 31, nt = (e2 >> 9) % NT, s = e2 / 3584;
    int k = s * 32 + kmapB(ln, j), n = nt * 16 + (ln & 15);
    v = (k < 100 && n < 100) ? W2[k * 100 + n] : 0.f;
  } else if (e < W4F_OFF) {                         // W3f
    int e2 = e - W3F_OFF;
    int j = e2 & 15, ln = (e2 >> 4) & 31, nt = (e2 >> 9) % NT, s = e2 / 3584;
    int k = s * 32 + kmapB(ln, j), n = nt * 16 + (ln & 15);
    v = (k < 100 && n < 100) ? W3[k * 100 + n] : 0.f;
  } else {                                          // W4f [s][lane][16], col 0 only
    int e2 = e - W4F_OFF;
    int j = e2 & 15, ln = (e2 >> 4) & 31, s = e2 >> 9;
    int k = s * 32 + kmapB(ln, j), n = ln & 15;
    v = (n == 0 && k < 100) ? W4[k] : 0.f;
  }
  wblob[e] = (_Float16)v;
  if (e < BIAS_FLOATS) {                            // bias[3][112], zero-padded
    int q = e / 112, c = e % 112;
    const float* bsrc = (q == 0) ? b1 : (q == 1) ? b2 : b3;
    biasbuf[e] = (c < 100) ? bsrc[c] : 0.f;
  }
}

// ------------------------------------------------------ fused WMMA MLP ------
__device__ __forceinline__ void hidden_layer(const _Float16* __restrict__ Wf,
                                             const float* __restrict__ bias,
                                             _Float16 (*H)[HSTRIDE], int lane) {
  v8f zero = {};
  v8f acc[NT];
#pragma unroll
  for (int nt = 0; nt < NT; ++nt) acc[nt] = zero;
  const _Float16* hrow = H[lane & 15];
#pragma unroll
  for (int s = 0; s < KS; ++s) {
    v16h a;
    int k0 = s * 32 + ((lane < 16) ? 0 : 8);
    int k1 = s * 32 + ((lane < 16) ? 16 : 24);
#pragma unroll
    for (int j = 0; j < 8; ++j) a[j] = hrow[k0 + j];
#pragma unroll
    for (int j = 0; j < 8; ++j) a[8 + j] = hrow[k1 + j];
#pragma unroll
    for (int nt = 0; nt < NT; ++nt) {
      v16h bf = *(const v16h*)(Wf + (((s * NT + nt) * 32 + lane) << 4));
      acc[nt] = wmma16(a, bf, acc[nt]);
    }
  }
  int col0 = lane & 15;
  int roff = (lane < 16) ? 0 : 8;
#pragma unroll
  for (int nt = 0; nt < NT; ++nt) {
    float bc = bias[nt * 16 + col0];
#pragma unroll
    for (int i = 0; i < 8; ++i) {
      float h = acc[nt][i] + bc;
      h = h > 0.f ? h : 0.f;
      H[i + roff][nt * 16 + col0] = (_Float16)h;
    }
  }
}

__global__ __launch_bounds__(256, 1)
void mlp_wmma_kernel(const float* __restrict__ w, const float* __restrict__ matched,
                     const float* __restrict__ hist_sum, const float* __restrict__ hist_sum_sq,
                     const float* __restrict__ hist_deg, const float* __restrict__ size_,
                     const float* __restrict__ sum_sol_sq, const float* __restrict__ num_skip,
                     const float* __restrict__ max_sol, const float* __restrict__ min_sol,
                     const float* __restrict__ b4, const float* __restrict__ mean_w,
                     const _Float16* __restrict__ wblob, const float* __restrict__ biasbuf,
                     float* __restrict__ logits) {
  __shared__ __attribute__((aligned(32))) _Float16 Wblob_s[WBLOB_HALFS];
  __shared__ float bias_s[BIAS_FLOATS];
  __shared__ __attribute__((aligned(32))) _Float16 Hb[WAVES][16][HSTRIDE];

  const int tid = threadIdx.x;

  // ---- stage fragment-ordered weight blob into LDS (68,608 B) ----
#if __has_builtin(__builtin_amdgcn_global_load_async_to_lds_b128)
  {
    typedef __attribute__((address_space(1))) v4i* gp_t;   // global (device) AS
    typedef __attribute__((address_space(3))) v4i* lp_t;   // LDS (shared) AS
    const v4i* gsrc = (const v4i*)wblob;
    v4i* ldst = (v4i*)&Wblob_s[0];
    for (int e = tid; e < WBLOB_B128; e += 256)
      __builtin_amdgcn_global_load_async_to_lds_b128((gp_t)(gsrc + e),
                                                     (lp_t)(ldst + e), 0, 0);
    asm volatile("s_wait_asynccnt 0" ::: "memory");
  }
#else
  for (int e = tid; e < WBLOB_B128; e += 256)
    ((uint4*)&Wblob_s[0])[e] = ((const uint4*)wblob)[e];
#endif
  for (int n = tid; n < BIAS_FLOATS; n += 256) bias_s[n] = biasbuf[n];

  const _Float16* W1f = Wblob_s + W1F_OFF;
  const _Float16* W2f = Wblob_s + W2F_OFF;
  const _Float16* W3f = Wblob_s + W3F_OFF;
  const _Float16* W4f = Wblob_s + W4F_OFF;
  const float* bs1 = bias_s;
  const float* bs2 = bias_s + 112;
  const float* bs3 = bias_s + 224;

  const int wv = tid >> 5, lane = tid & 31;
  // zero per-wave H buffer once (cols 112..127 must read as 0 for K padding)
  for (int e = lane; e < 16 * HSTRIDE; e += 32)
    (&Hb[wv][0][0])[e] = (_Float16)0.f;
  __syncthreads();

  // ---- features for this wave's 16 rows (lane pairs L,L+16 share a row) ----
  const int M = lane & 15;
  const int rowbase = (blockIdx.x * WAVES + wv) * 16;
  const int r  = rowbase + M;
  const int u  = r % U1C;
  const int bt = r / U1C;
  const int t  = bt % TC;
  const float fi = (float)(t + 1);
  const float inv_i = 1.f / fi;
  float f0 = w[r], f1 = matched[r];
  float hs = hist_sum[r], hss = hist_sum_sq[r], hd = hist_deg[r];
  float sz = size_[bt], ssq = sum_sol_sq[bt], nsk = num_skip[bt];
  float f2 = mean_w[bt];
  float f3 = (u == 0) ? -1.f : hs * inv_i;
  float f4 = (u == 0) ? -1.f : (hss - hs * hs * inv_i) * inv_i;
  float f5 = (u == 0) ? -1.f : hd * inv_i;
  float f6 = fi * (1.f / (float)TC);
  float f7 = sz * (1.f / (float)(U1C - 1));
  float curr = fi - nsk;
  float invc = 1.f / curr;
  float f8 = sz * invc;
  float f9 = (ssq - sz * sz * invc) * invc;
  float f10 = nsk * inv_i;
  float f11 = max_sol[bt], f12 = min_sol[bt];

  // A-fragment for layer 1 (K=13 padded to 32, built directly in registers)
  v16h a1 = {};
  if (lane < 16) {   // K = 0..7 in j0..7 ; K=16..23 (zero pad) in j8..15
    a1[0] = (_Float16)f0;  a1[1] = (_Float16)f1;  a1[2] = (_Float16)f2;
    a1[3] = (_Float16)f3;  a1[4] = (_Float16)f4;  a1[5] = (_Float16)f5;
    a1[6] = (_Float16)f6;  a1[7] = (_Float16)f7;
  } else {           // K = 8..15 in j0..7 -> features 8..12 then zeros
    a1[0] = (_Float16)f8;  a1[1] = (_Float16)f9;  a1[2] = (_Float16)f10;
    a1[3] = (_Float16)f11; a1[4] = (_Float16)f12;
  }

  // ---- layer 1: 7 WMMAs ----
  {
    v8f zero = {};
    v8f acc[NT];
#pragma unroll
    for (int nt = 0; nt < NT; ++nt) acc[nt] = zero;
#pragma unroll
    for (int nt = 0; nt < NT; ++nt) {
      v16h bf = *(const v16h*)(W1f + ((nt * 32 + lane) << 4));
      acc[nt] = wmma16(a1, bf, acc[nt]);
    }
    int col0 = lane & 15, roff = (lane < 16) ? 0 : 8;
#pragma unroll
    for (int nt = 0; nt < NT; ++nt) {
      float bc = bs1[nt * 16 + col0];
#pragma unroll
      for (int i = 0; i < 8; ++i) {
        float h = acc[nt][i] + bc;
        h = h > 0.f ? h : 0.f;
        Hb[wv][i + roff][nt * 16 + col0] = (_Float16)h;
      }
    }
  }

  // ---- layers 2 and 3: 28 WMMAs each ----
  hidden_layer(W2f, bs2, Hb[wv], lane);
  hidden_layer(W3f, bs3, Hb[wv], lane);

  // ---- layer 4: 4 WMMAs, only column 0 meaningful ----
  {
    v8f o = {};
    const _Float16* hrow = Hb[wv][M];
#pragma unroll
    for (int s = 0; s < KS; ++s) {
      v16h a;
      int k0 = s * 32 + ((lane < 16) ? 0 : 8);
      int k1 = s * 32 + ((lane < 16) ? 16 : 24);
#pragma unroll
      for (int j = 0; j < 8; ++j) a[j] = hrow[k0 + j];
#pragma unroll
      for (int j = 0; j < 8; ++j) a[8 + j] = hrow[k1 + j];
      v16h bf = *(const v16h*)(W4f + ((s * 32 + lane) << 4));
      o = wmma16(a, bf, o);
    }
    if ((lane & 15) == 0) {      // lanes 0 (rows 0..7) and 16 (rows 8..15)
      float bias4 = b4[0];
      int off = (lane < 16) ? 0 : 8;
#pragma unroll
      for (int i = 0; i < 8; ++i)
        logits[rowbase + off + i] = o[i] + bias4;
    }
  }
}

// ------------------------------------------- masked log-softmax per (b,t) ---
__global__ __launch_bounds__(128)
void softmax_kernel(const float* logits_in, const unsigned char* __restrict__ mask,
                    float* log_p, float* __restrict__ sel_out,
                    float* __restrict__ ll_part, float* __restrict__ ent_part) {
  __shared__ float rv[128];
  __shared__ int   ri[128];
  const int row = blockIdx.x, tid = threadIdx.x;
  const bool valid = tid < U1C;
  float x = -3.4e38f;
  if (valid) {
    float lg = logits_in[(size_t)row * U1C + tid];
    x = mask[(size_t)row * U1C + tid] ? -1.0e6f : lg;
  }
  rv[tid] = x;
  ri[tid] = valid ? tid : (1 << 30);
  __syncthreads();
  for (int off = 64; off > 0; off >>= 1) {          // argmax, first-index ties
    if (tid < off) {
      float bv = rv[tid + off]; int bi = ri[tid + off];
      if (bv > rv[tid] || (bv == rv[tid] && bi < ri[tid])) { rv[tid] = bv; ri[tid] = bi; }
    }
    __syncthreads();
  }
  const float vmax = rv[0];
  const int   sel  = ri[0];
  __syncthreads();
  float e = valid ? expf(x - vmax) : 0.f;
  rv[tid] = e; __syncthreads();
  for (int off = 64; off > 0; off >>= 1) {
    if (tid < off) rv[tid] += rv[tid + off];
    __syncthreads();
  }
  const float logZ = vmax + logf(rv[0]);
  __syncthreads();
  float lp = x - logZ;
  float ent = 0.f;
  if (valid) {
    log_p[(size_t)row * U1C + tid] = lp;            // in-place over logits
    ent = -lp * expf(lp);
  }
  rv[tid] = ent; __syncthreads();
  for (int off = 64; off > 0; off >>= 1) {
    if (tid < off) rv[tid] += rv[tid + off];
    __syncthreads();
  }
  if (tid == 0) {
    sel_out[row]  = (float)sel;
    ll_part[row]  = vmax - logZ;                    // log_p at argmax
    ent_part[row] = rv[0];
  }
}

// ----------------------------------------------- per-batch reductions -------
__global__ __launch_bounds__(256)
void reduce_bt_kernel(const float* __restrict__ ll_part, const float* __restrict__ ent_part,
                      float* __restrict__ ll_out, float* __restrict__ ent_b) {
  __shared__ float s1[256], s2[256];
  const int b = blockIdx.x, tid = threadIdx.x;
  float a = 0.f, c = 0.f;
  if (tid < TC) { a = ll_part[b * TC + tid]; c = ent_part[b * TC + tid]; }
  s1[tid] = a; s2[tid] = c; __syncthreads();
  for (int off = 128; off > 0; off >>= 1) {
    if (tid < off) { s1[tid] += s1[tid + off]; s2[tid] += s2[tid + off]; }
    __syncthreads();
  }
  if (tid == 0) { ll_out[b] = s1[0]; ent_b[b] = s2[0]; }
}

__global__ __launch_bounds__(64)
void final_kernel(const float* __restrict__ ent_b, float* __restrict__ ent_out) {
  __shared__ float s[64];
  int tid = threadIdx.x;
  s[tid] = ent_b[tid];
  __syncthreads();
  for (int off = 32; off > 0; off >>= 1) {
    if (tid < off) s[tid] += s[tid + off];
    __syncthreads();
  }
  if (tid == 0) ent_out[0] = s[0] * (1.f / (float)BC);
}

// ---------------------------------------------------------------------------
extern "C" void kernel_launch(void* const* d_in, const int* in_sizes, int n_in,
                              void* d_out, int out_size, void* d_ws, size_t ws_size,
                              hipStream_t stream) {
  const float* w           = (const float*)d_in[0];
  const float* matched     = (const float*)d_in[1];
  const float* hist_sum    = (const float*)d_in[2];
  const float* hist_sum_sq = (const float*)d_in[3];
  const float* hist_deg    = (const float*)d_in[4];
  const float* size_       = (const float*)d_in[5];
  const float* sum_sol_sq  = (const float*)d_in[6];
  const float* num_skip    = (const float*)d_in[7];
  const float* max_sol     = (const float*)d_in[8];
  const float* min_sol     = (const float*)d_in[9];
  const float* W1          = (const float*)d_in[10];
  const float* b1          = (const float*)d_in[11];
  const float* W2          = (const float*)d_in[12];
  const float* b2          = (const float*)d_in[13];
  const float* W3          = (const float*)d_in[14];
  const float* b3          = (const float*)d_in[15];
  const float* W4          = (const float*)d_in[16];
  const float* b4          = (const float*)d_in[17];
  const unsigned char* mask = (const unsigned char*)d_in[18];

  float* out    = (float*)d_out;
  float* logits = out;                       // [NROWS] log_p region (scratch, then final)
  float* sel    = out + NROWS;               // [BTC]
  float* ll     = out + NROWS + BTC;         // [BC]
  float* ent    = out + NROWS + BTC + BC;    // [1]

  float* ws       = (float*)d_ws;
  float* mw       = ws;                      // [BTC]
  float* ll_part  = ws + BTC;                // [BTC]
  float* ent_part = ws + 2 * BTC;            // [BTC]
  float* ent_b    = ws + 3 * BTC;            // [BC]
  float* biasbuf  = ws + 3 * BTC + BC;       // [336]
  _Float16* wblob = (_Float16*)(ws + 3 * BTC + BC + BIAS_FLOATS); // 16B-aligned blob

  prep_weights_kernel<<<WBLOB_HALFS / 256, 256, 0, stream>>>(
      W1, W2, W3, W4, b1, b2, b3, wblob, biasbuf);
  meanw_kernel<<<BTC / 8, 256, 0, stream>>>(w, mw);
  mlp_wmma_kernel<<<NROWS / (WAVES * 16), 256, 0, stream>>>(
      w, matched, hist_sum, hist_sum_sq, hist_deg, size_, sum_sol_sq, num_skip,
      max_sol, min_sol, b4, mw, wblob, biasbuf, logits);
  softmax_kernel<<<BTC, 128, 0, stream>>>(logits, mask, logits, sel, ll_part, ent_part);
  reduce_bt_kernel<<<BC, 256, 0, stream>>>(ll_part, ent_part, ll, ent_b);
  final_kernel<<<1, 64, 0, stream>>>(ent_b, ent);
}
